// DualMemoryLayer_53618371723295
// MI455X (gfx1250) — matrix operand order repaired and checked by reading, not compiled, not run on hardware
//
#include <hip/hip_runtime.h>
#include <hip/hip_bf16.h>

// ---------------------------------------------------------------------------
// E25 entmax dual-memory recurrence for MI455X (gfx1250, wave32, WMMA).
//   B=8, T=1024, D=512, N_SLOTS=32.
// Strategy:
//   * Big GEMMs (x@W_in^T, x_proj@W_x^T, gated@W_out^T) via v_wmma_f32_16x16x32_bf16,
//     2 N-tiles per wave with independent accumulators (breaks WMMA RAW chain,
//     reuses the A fragment), WGP-scope prefetch of the K-stream.
//   * Sequential scan: 1 workgroup per batch, tape[32][512] f32 resident in LDS,
//     per-step GEMVs packed into WMMA tiles (M=1 row), dual-accumulator
//     interleave, bias/xw fused into the GEMV epilogue (8 barriers/step).
//   * entmax-1.5 fully in-register in one wave: rank via shfl, descending sort
//     realized as a ds_permute_b32 scatter-by-rank, 5-step prefix scans,
//     ballot/popcount support selection.
// ---------------------------------------------------------------------------

typedef __attribute__((ext_vector_type(16))) __bf16 v16bf;
typedef __attribute__((ext_vector_type(8)))  __bf16 bfx8;
typedef __attribute__((ext_vector_type(8)))  float  v8f;

#define DEV __device__ __forceinline__

constexpr int B_    = 8;
constexpr int T_    = 1024;
constexpr int D_    = 512;
constexpr int NSLOT = 32;

// ---- bf16 helpers (RNE) ----------------------------------------------------
DEV __bf16 f2bf(float f) {
  unsigned u = __builtin_bit_cast(unsigned, f);
  u += 0x7FFFu + ((u >> 16) & 1u);
  unsigned short h = (unsigned short)(u >> 16);
  return __builtin_bit_cast(__bf16, h);
}
DEV float bf2f(__bf16 b) {
  unsigned short h = __builtin_bit_cast(unsigned short, b);
  unsigned u = ((unsigned)h) << 16;
  return __builtin_bit_cast(float, u);
}
DEV bfx8 bf8zero() {
  __bf16 z = __builtin_bit_cast(__bf16, (unsigned short)0);
  bfx8 v = {z, z, z, z, z, z, z, z};
  return v;
}
DEV v16bf cat16(bfx8 lo, bfx8 hi) {
  return __builtin_shufflevector(lo, hi, 0,1,2,3,4,5,6,7,8,9,10,11,12,13,14,15);
}

// ---- f32 -> bf16 conversion ------------------------------------------------
__global__ void __launch_bounds__(256)
cvt_f32_to_bf16(const float* __restrict__ src, __bf16* __restrict__ dst, int n) {
  int i = blockIdx.x * blockDim.x + threadIdx.x;
  if (i < n) dst[i] = f2bf(src[i]);
}

// ---- generic C = A[M,K] * B[N,K]^T ----------------------------------------
// One M16 x N32 super-tile per wave: A fragment loaded once per k-step and fed
// to TWO independent-accumulator WMMAs (pipelines the XDL, halves A traffic).
// Operand layouts used here (16-bit, 16x16x32; contiguous k, same for A and B):
//   A lane L: row = L&15, g = L>>4; elems 0..15 = K[16g..16g+15]
//   B lane L: col = L&15, g = L>>4; elems 0..15 = K[16g..16g+15] (row-major W)
//   D vgpr r: row = r + 8*(L>>4), col = L&15
// mode 0: store f32; mode 1: store bf16(silu(x))
__global__ void __launch_bounds__(256)
gemm_bf16_wmma(const __bf16* __restrict__ A, int lda,
               const __bf16* __restrict__ Bm, int ldb,
               void* __restrict__ Cv, int ldc,
               int M, int N, int K, int mode)
{
  const int lane    = threadIdx.x & 31;
  const int wave    = threadIdx.x >> 5;
  const int tilesN2 = N >> 5;                        // super-tiles along N
  const long long tile = (long long)blockIdx.x * (blockDim.x >> 5) + wave;
  const long long mt  = tile / tilesN2;
  const long long nt0 = (tile % tilesN2) * 2;
  const long long nt1 = nt0 + 1;
  if (mt >= (M >> 4)) return;                        // wave-uniform guard
  const int row = lane & 15;
  const int g   = lane >> 4;

  const __bf16* Ap  = A  + (mt  * 16 + row) * (long long)lda;
  const __bf16* Bp0 = Bm + (nt0 * 16 + row) * (long long)ldb;
  const __bf16* Bp1 = Bp0 + 16 * (long long)ldb;

  v8f c0 = {0.f, 0.f, 0.f, 0.f, 0.f, 0.f, 0.f, 0.f};
  v8f c1 = {0.f, 0.f, 0.f, 0.f, 0.f, 0.f, 0.f, 0.f};
  for (int k = 0; k < K; k += 32) {
    if (k + 32 < K) {                                // WGP-scope prefetch
      __builtin_prefetch(Ap  + k + 32, 0, 3);
      __builtin_prefetch(Bp0 + k + 32, 0, 3);
      __builtin_prefetch(Bp1 + k + 32, 0, 3);
    }
    bfx8 a0 = *(const bfx8*)(Ap + k + 16 * g);
    bfx8 a1 = *(const bfx8*)(Ap + k + 16 * g + 8);
    v16bf a = cat16(a0, a1);
    v16bf b0 = cat16(*(const bfx8*)(Bp0 + k + 16 * g),
                     *(const bfx8*)(Bp0 + k + 16 * g + 8));
    v16bf b1 = cat16(*(const bfx8*)(Bp1 + k + 16 * g),
                     *(const bfx8*)(Bp1 + k + 16 * g + 8));
    c0 = __builtin_amdgcn_wmma_f32_16x16x32_bf16(false, a, false, b0, (short)0, c0,
                                                 false, false);
    c1 = __builtin_amdgcn_wmma_f32_16x16x32_bf16(false, a, false, b1, (short)0, c1,
                                                 false, false);
  }

  const long long col0 = nt0 * 16 + row;
  const long long col1 = nt1 * 16 + row;
  if (mode == 0) {
    float* C = (float*)Cv;
#pragma unroll
    for (int r = 0; r < 8; ++r) {
      long long mr = mt * 16 + r + 8 * g;
      C[mr * (long long)ldc + col0] = c0[r];
      C[mr * (long long)ldc + col1] = c1[r];
    }
  } else {
    __bf16* C = (__bf16*)Cv;
#pragma unroll
    for (int r = 0; r < 8; ++r) {
      long long mr = mt * 16 + r + 8 * g;
      float v0 = c0[r], v1 = c1[r];
      C[mr * (long long)ldc + col0] = f2bf(v0 / (1.f + __expf(-v0)));  // silu
      C[mr * (long long)ldc + col1] = f2bf(v1 / (1.f + __expf(-v1)));
    }
  }
}

// ---- in-scan GEMV: out[512] = hb[512] @ W[512,512]^T via WMMA --------------
// 16 waves x (2 N-tiles, dual accumulators); M=1 packed into row 0 of the A
// tile. Optional fused epilogue adds add0[n] + add1[n] (xw_t and b_h).
DEV void gemv16_wmma(const __bf16* __restrict__ W, const __bf16* hb,
                     float* out, int tid,
                     const float* __restrict__ add0,
                     const float* __restrict__ add1)
{
  const int lane = tid & 31;
  const int wave = tid >> 5;
  const int row  = lane & 15;
  const int g    = lane >> 4;
  const int nt0  = wave * 2;
  const int nt1  = nt0 + 1;
  const __bf16* Wp0 = W + (nt0 * 16 + row) * (long long)D_;
  const __bf16* Wp1 = Wp0 + 16 * (long long)D_;

  v8f c0 = {0.f, 0.f, 0.f, 0.f, 0.f, 0.f, 0.f, 0.f};
  v8f c1 = {0.f, 0.f, 0.f, 0.f, 0.f, 0.f, 0.f, 0.f};
  for (int k = 0; k < D_; k += 32) {
    bfx8 a0 = bf8zero(), a1 = bf8zero();
    if (row == 0) {                                  // only M=0 row is live
      a0 = *(const bfx8*)(hb + k + 16 * g);          // ds_load_b128
      a1 = *(const bfx8*)(hb + k + 16 * g + 8);
    }
    v16bf a  = cat16(a0, a1);
    v16bf b0 = cat16(*(const bfx8*)(Wp0 + k + 16 * g),
                     *(const bfx8*)(Wp0 + k + 16 * g + 8));
    v16bf b1 = cat16(*(const bfx8*)(Wp1 + k + 16 * g),
                     *(const bfx8*)(Wp1 + k + 16 * g + 8));
    c0 = __builtin_amdgcn_wmma_f32_16x16x32_bf16(false, a, false, b0, (short)0, c0,
                                                 false, false);
    c1 = __builtin_amdgcn_wmma_f32_16x16x32_bf16(false, a, false, b1, (short)0, c1,
                                                 false, false);
  }
  if (lane < 16) {                                   // D row 0 = vgpr0, lanes 0..15
    int n0 = nt0 * 16 + lane;
    int n1 = nt1 * 16 + lane;
    float v0 = c0[0], v1 = c1[0];
    if (add0) { v0 += add0[n0]; v1 += add0[n1]; }
    if (add1) { v0 += add1[n0]; v1 += add1[n1]; }
    out[n0] = v0;
    out[n1] = v1;
  }
}

// ---- exact 1.5-entmax over 32 lanes, fully in-register ---------------------
DEV float entmax15_lane(float z, int lane)
{
  float x = 0.5f * z;
  float m = x;                                       // max-reduce
#pragma unroll
  for (int mask = 16; mask >= 1; mask >>= 1) m = fmaxf(m, __shfl_xor(m, mask, 32));
  x -= m;

  int rank = 0;                                      // descending rank (stable)
  for (int j = 0; j < 32; ++j) {
    float xj = __shfl(x, j, 32);
    rank += (xj > x) || (xj == x && j < lane);
  }
  // sorted-descending value at each lane: scatter by rank (ds_permute_b32)
  float zs = __builtin_bit_cast(float,
      __builtin_amdgcn_ds_permute(rank << 2, __builtin_bit_cast(int, x)));

  float cs = zs, css = zs * zs;                      // inclusive prefix scans
#pragma unroll
  for (int off = 1; off < 32; off <<= 1) {
    float u = __shfl_up(cs, off, 32);
    float v = __shfl_up(css, off, 32);
    if (lane >= off) { cs += u; css += v; }
  }
  float k1    = (float)(lane + 1);
  float mean  = cs / k1;
  float msq   = css / k1;
  float ssv   = k1 * (msq - mean * mean);
  float delta = (1.f - ssv) / k1;
  float tau   = mean - sqrtf(fmaxf(delta, 0.f));
  unsigned long long bal = __ballot(tau <= zs);
  int support = __popcll(bal) - 1;
  float tau_star = __shfl(tau, support, 32);
  float p = fmaxf(x - tau_star, 0.f);
  return p * p;
}

// ---- sequential dual-memory scan: 1 workgroup per batch --------------------
__global__ void __launch_bounds__(512)
scan_kernel(const __bf16* __restrict__ Wh, const __bf16* __restrict__ Ww,
            const float* __restrict__ xw, const float* __restrict__ bh,
            float* __restrict__ hs, float* __restrict__ tape_out,
            float* __restrict__ work_out)
{
  extern __shared__ char smem[];
  char* sp = smem;
  float*  tape = (float*)sp;  sp += NSLOT * D_ * 4;  // 64 KB, LDS-resident tape
  float*  hpre = (float*)sp;  sp += D_ * 4;
  float*  hnew = (float*)sp;  sp += D_ * 4;
  float*  wval = (float*)sp;  sp += D_ * 4;
  __bf16* hb   = (__bf16*)sp; sp += D_ * 2;          // 16B-aligned
  float*  avec = (float*)sp;  sp += NSLOT * 4;
  float*  bvec = (float*)sp;  sp += NSLOT * 4;
  float*  scor = (float*)sp;  sp += NSLOT * 4;

  const int tid = threadIdx.x;
  const int b   = blockIdx.x;
  const float scale = 0.04419417382415922f;          // 1/sqrt(512)

  for (int i = tid; i < NSLOT * D_; i += 512) tape[i] = 0.f;
  hb[tid]   = __builtin_bit_cast(__bf16, (unsigned short)0);
  hnew[tid] = 0.f;
  __syncthreads();

  const float* xwB = xw + (long long)b * T_ * D_;
  float*       hsB = hs + (long long)b * T_ * D_;

  for (int t = 0; t < T_; ++t) {
    // h_pre = work @ Wh^T + xw_t + b_h  (WMMA, fused epilogue)
    gemv16_wmma(Wh, hb, hpre, tid, xwB + t * D_, bh);
    __syncthreads();

    // r_scores[n] = <tape[n], h_pre> * scale   (16 lanes per slot)
    {
      const int slot = tid >> 4, l = tid & 15;
      const float* tp = tape + slot * D_;
      float s = 0.f;
      for (int i = l; i < D_; i += 16) s += tp[i] * hpre[i];
      s += __shfl_xor(s, 8, 32); s += __shfl_xor(s, 4, 32);
      s += __shfl_xor(s, 2, 32); s += __shfl_xor(s, 1, 32);
      if (l == 0) scor[slot] = s * scale;
    }
    __syncthreads();
    if (tid < 32) avec[tid] = entmax15_lane(scor[tid], tid);
    __syncthreads();

    // read + tanh -> h_new
    {
      float r = 0.f;
#pragma unroll 8
      for (int n = 0; n < NSLOT; ++n) r += avec[n] * tape[n * D_ + tid];
      float hv = tanhf(hpre[tid] + r);
      hnew[tid] = hv;
      hb[tid]   = f2bf(hv);
      hsB[t * D_ + tid] = hv;
    }
    __syncthreads();

    // w_val = h_new @ Ww^T  (WMMA)
    gemv16_wmma(Ww, hb, wval, tid, nullptr, nullptr);
    __syncthreads();

    // w_scores[n] = <tape[n], h_new> * scale
    {
      const int slot = tid >> 4, l = tid & 15;
      const float* tp = tape + slot * D_;
      float s = 0.f;
      for (int i = l; i < D_; i += 16) s += tp[i] * hnew[i];
      s += __shfl_xor(s, 8, 32); s += __shfl_xor(s, 4, 32);
      s += __shfl_xor(s, 2, 32); s += __shfl_xor(s, 1, 32);
      if (l == 0) scor[slot] = s * scale;
    }
    __syncthreads();
    if (tid < 32) bvec[tid] = entmax15_lane(scor[tid], tid);
    __syncthreads();

    // tape <- tape + b * (w_val - tape)
    {
      float wv = wval[tid];
#pragma unroll 8
      for (int n = 0; n < NSLOT; ++n) {
        float tp = tape[n * D_ + tid];
        tape[n * D_ + tid] = tp + bvec[n] * (wv - tp);
      }
    }
    __syncthreads();
  }

  for (int i = tid; i < NSLOT * D_; i += 512)
    tape_out[(long long)b * NSLOT * D_ + i] = tape[i];
  work_out[b * D_ + tid] = hnew[tid];
}

// ---- gating: gated = bf16(hs * silu(z)) ------------------------------------
__global__ void __launch_bounds__(256)
gate_kernel(const float* __restrict__ hs, const __bf16* __restrict__ sz,
            __bf16* __restrict__ gated, int total)
{
  int i = blockIdx.x * blockDim.x + threadIdx.x;
  if (i >= total) return;
  int r = i >> 9;                                    // / 512
  int c = i & 511;
  float sv = bf2f(sz[(long long)r * (2 * D_) + D_ + c]);   // silu(z) half
  gated[i] = f2bf(hs[i] * sv);
}

// ---------------------------------------------------------------------------
extern "C" void kernel_launch(void* const* d_in, const int* in_sizes, int n_in,
                              void* d_out, int out_size, void* d_ws, size_t ws_size,
                              hipStream_t stream)
{
  (void)in_sizes; (void)n_in; (void)out_size; (void)ws_size;
  const float* x     = (const float*)d_in[0];   // [8,1024,512]
  const float* W_in  = (const float*)d_in[1];   // [1024,512]
  const float* W_out = (const float*)d_in[2];   // [512,512]
  const float* W_h   = (const float*)d_in[3];   // [512,512]
  const float* W_x   = (const float*)d_in[4];   // [512,512]
  const float* b_h   = (const float*)d_in[5];   // [512]
  const float* W_wr  = (const float*)d_in[6];   // [512,512]

  const int BT = B_ * T_;                       // 8192

  // workspace partition (256B aligned)
  char* p = (char*)d_ws;
  auto take = [&](size_t bytes) -> char* {
    char* r = p; p += (bytes + 255) & ~(size_t)255; return r;
  };
  __bf16* xb     = (__bf16*)take((size_t)BT * D_ * 2);
  __bf16* Wb_in  = (__bf16*)take((size_t)2 * D_ * D_ * 2);
  __bf16* Wb_x   = (__bf16*)take((size_t)D_ * D_ * 2);
  __bf16* Wb_h   = (__bf16*)take((size_t)D_ * D_ * 2);
  __bf16* Wb_wr  = (__bf16*)take((size_t)D_ * D_ * 2);
  __bf16* Wb_out = (__bf16*)take((size_t)D_ * D_ * 2);
  __bf16* szb    = (__bf16*)take((size_t)BT * 2 * D_ * 2); // silu(x@W_in^T) bf16 [8192,1024]
  float*  xwf    = (float*) take((size_t)BT * D_ * 4);     // x_proj @ W_x^T
  float*  hsf    = (float*) take((size_t)BT * D_ * 4);     // scan outputs h_t
  __bf16* gated  = (__bf16*)take((size_t)BT * D_ * 2);

  // 1) f32 -> bf16 conversions
  auto cvt = [&](const float* s, __bf16* d, int n) {
    cvt_f32_to_bf16<<<(n + 255) / 256, 256, 0, stream>>>(s, d, n);
  };
  cvt(x,     xb,     BT * D_);
  cvt(W_in,  Wb_in,  2 * D_ * D_);
  cvt(W_x,   Wb_x,   D_ * D_);
  cvt(W_h,   Wb_h,   D_ * D_);
  cvt(W_wr,  Wb_wr,  D_ * D_);
  cvt(W_out, Wb_out, D_ * D_);

  // 2) szb = silu(x @ W_in^T)   [8192,1024] bf16 (both halves silu'd;
  //    cols 0..511 = x_proj, cols 512..1023 = silu(z) for gating)
  {
    int tiles = (BT / 16) * ((2 * D_) / 32);    // 16384 super-tiles
    gemm_bf16_wmma<<<tiles / 8, 256, 0, stream>>>(
        xb, D_, Wb_in, D_, szb, 2 * D_, BT, 2 * D_, D_, /*mode=*/1);
  }
  // 3) xw = x_proj @ W_x^T   (A = szb cols 0..511, lda = 1024)
  {
    int tiles = (BT / 16) * (D_ / 32);          // 8192 super-tiles
    gemm_bf16_wmma<<<tiles / 8, 256, 0, stream>>>(
        szb, 2 * D_, Wb_x, D_, xwf, D_, BT, D_, D_, /*mode=*/0);
  }
  // 4) sequential scan: 1 WG per batch, tape in LDS, per-step WMMA GEMVs
  {
    float* tape_out = (float*)d_out + (size_t)BT * D_;
    float* work_out = tape_out + (size_t)B_ * NSLOT * D_;
    size_t smem = (size_t)NSLOT * D_ * 4 + 3 * D_ * 4 + D_ * 2 + 3 * NSLOT * 4;
    scan_kernel<<<B_, 512, smem, stream>>>(Wb_h, Wb_wr, xwf, b_h, hsf,
                                           tape_out, work_out);
  }
  // 5) gated = bf16(hs * silu(z))
  gate_kernel<<<(BT * D_ + 255) / 256, 256, 0, stream>>>(hsf, szb, gated, BT * D_);

  // 6) out = gated @ W_out^T  -> d_out (f32)
  {
    int tiles = (BT / 16) * (D_ / 32);
    gemm_bf16_wmma<<<tiles / 8, 256, 0, stream>>>(
        gated, D_, Wb_out, D_, (float*)d_out, D_, BT, D_, D_, /*mode=*/0);
  }
}
